// Model_41583873360247
// MI455X (gfx1250) — matrix operand order repaired and checked by reading, not compiled, not run on hardware
//
#include <hip/hip_runtime.h>

#define B_    512
#define T_    2000
#define DIN   85
#define H_    256
#define DOUT  33
#define ALPHA 0.2f

typedef _Float16 v16h __attribute__((ext_vector_type(16)));
typedef _Float16 v8h  __attribute__((ext_vector_type(8)));
typedef float    v8f  __attribute__((ext_vector_type(8)));

#define HSTR 264   // h-stage row stride in halfs (256 + 8 pad, 16B-multiple)
#define XSTR 104   // x-stage row stride in halfs (96 + 8 pad, 16B-multiple)

#define WMMA_F16(a, b, c) \
    __builtin_amdgcn_wmma_f32_16x16x32_f16(false, (a), false, (b), (short)0, (c), false, false)

// Split workgroup barrier (CDNA5 S_BARRIER_SIGNAL/WAIT, barrier id -1).
// Signal only after our LDS stores are globally visible (drain DScnt).
__device__ __forceinline__ void wg_barrier_signal_drained() {
    asm volatile("s_wait_dscnt 0x0\n\ts_barrier_signal -1" ::: "memory");
}
__device__ __forceinline__ void wg_barrier_wait() {
    asm volatile("s_barrier_wait -1" ::: "memory");
}

// Build a 16x32 f16 A fragment from an LDS tile (layout: lane%16 = M,
// lane/16 selects K-half; elems 0..15 = K[kh*16..kh*16+15]).
__device__ __forceinline__ v16h load_a_frag(const _Float16* base, int stride,
                                            int kofs, int m, int kh) {
    const _Float16* p = base + m * stride + kofs + kh * 16;
    v8h lo = *(const v8h*)(p);
    v8h hi = *(const v8h*)(p + 8);
    v16h a;
#pragma unroll
    for (int i = 0; i < 8; ++i) { a[i] = lo[i]; a[i + 8] = hi[i]; }
    return a;
}

__device__ __forceinline__ float softplus_f(float v) {
    return fmaxf(v, 0.0f) + __logf(1.0f + __expf(-fabsf(v)));
}
__device__ __forceinline__ float sigmoid_f(float v) {
    return 1.0f / (1.0f + __expf(-v));
}

__global__ __launch_bounds__(512, 1) void rnn_fused(
    const float* __restrict__ x, const float* __restrict__ y,
    const float* __restrict__ cmask, const float* __restrict__ weight,
    const float* __restrict__ bias, const float* __restrict__ w_out,
    const float* __restrict__ b_out, float* __restrict__ out,
    float* __restrict__ partials)
{
    __shared__ __align__(16) _Float16 hA[2][16 * HSTR];   // double-buffered h stage
    __shared__ __align__(16) _Float16 xA[2][16 * XSTR];   // double-buffered x stage
    __shared__ __align__(16) _Float16 boutL[96 * 8 * 16]; // pre-swizzled W_out frags, 24 KB
    __shared__ float red[512];

    const int tid  = threadIdx.x;
    const int wave = tid >> 5;       // 0..15, each owns 16 gate columns
    const int lane = tid & 31;
    const int lo16 = lane & 15;
    const int kh   = lane >> 4;
    const int b0   = blockIdx.x * 16;

    const int n = wave * 16 + lo16;  // this lane's gate column

    // ---- one-time B-fragment preload into VGPRs (B layout: lane%16 = N,
    // lanes 0-15 hold K=kb*32+0..15, lanes 16-31 hold K=kb*32+16..31) ----
    v16h Brec[8], Bin[3];
    const float bv = bias[n];
#pragma unroll
    for (int kb = 0; kb < 8; ++kb) {
        v16h f;
#pragma unroll
        for (int e = 0; e < 16; ++e) {
            int k = kb * 32 + kh * 16 + e;
            f[e] = (_Float16)weight[(size_t)(DIN + k) * H_ + n];
        }
        Brec[kb] = f;
    }
#pragma unroll
    for (int kb = 0; kb < 3; ++kb) {
        v16h f;
#pragma unroll
        for (int e = 0; e < 16; ++e) {
            int k = kb * 32 + kh * 16 + e;
            f[e] = (k < DIN) ? (_Float16)weight[(size_t)k * H_ + n] : (_Float16)0.0f;
        }
        Bin[kb] = f;
    }

    // Waves 0..2 own output N-tiles 0..47; W_out fragments live in a per-lane
    // swizzled LDS image (same-wave DS ordering makes them visible, no barrier).
    const bool outWave = (wave < 3);
    const int  no = wave * 16 + lo16;          // output column (valid if < 33)
    float bo = 0.0f;
    if (outWave) {
        bo = (no < DOUT) ? b_out[no] : 0.0f;
#pragma unroll
        for (int kb = 0; kb < 8; ++kb) {
            v16h f;
#pragma unroll
            for (int e = 0; e < 16; ++e) {
                int k = kb * 32 + kh * 16 + e;
                f[e] = (no < DOUT) ? (_Float16)w_out[(size_t)k * DOUT + no] : (_Float16)0.0f;
            }
            *(v16h*)&boutL[((size_t)tid * 8 + kb) * 16] = f;
        }
    }

    // x prefetch mapping: thread -> (row 0..15, 3 contiguous input features)
    const int xrow = tid >> 5;
    const int xd0  = (tid & 31) * 3;
    const float* xp = x + (size_t)(b0 + xrow) * T_ * DIN;  // running pointer (+= DIN/step)

    // running epilogue pointers (waves 0..2): advance by B_*DOUT per emitted step
    const size_t ebase = (size_t)(b0 + 8 * kh) * DOUT + no;
    const float* yp = y + ebase;
    const float* mp = cmask + ebase;
    float*       op = out + ebase;

    float hreg[8];
#pragma unroll
    for (int r = 0; r < 8; ++r) hreg[r] = 0.0f;

    float xr[3];
#pragma unroll
    for (int j = 0; j < 3; ++j) {
        int d = xd0 + j;
        xr[j] = (d < DIN) ? xp[d] : 0.0f;
    }

    float cost = 0.0f;

    for (int t = 0; t < T_; ++t) {
        _Float16* hS = hA[t & 1];
        _Float16* xS = xA[t & 1];

        // stage x_t and h_{t-1} (f16) into this step's buffer
#pragma unroll
        for (int j = 0; j < 3; ++j)
            xS[xrow * XSTR + xd0 + j] = (_Float16)xr[j];
#pragma unroll
        for (int r = 0; r < 8; ++r)
            hS[(r + 8 * kh) * HSTR + n] = (_Float16)hreg[r];

        wg_barrier_signal_drained();           // stores drained, arrive

        // ---- global prefetch inside the signal->wait window ----
        const bool doOut = outWave && (t > 0); // readout for s = t-1 uses hS = h_{t-1}
        float yv[8], mk[8];
        if (t + 1 < T_) {
#pragma unroll
            for (int j = 0; j < 3; ++j) {
                int d = xd0 + j;
                xr[j] = (d < DIN) ? xp[DIN + d] : 0.0f;
            }
        }
        if (doOut && no < DOUT) {
#pragma unroll
            for (int r = 0; r < 8; ++r) {
                yv[r] = yp[(size_t)r * DOUT];
                mk[r] = mp[(size_t)r * DOUT];
            }
        }
        xp += DIN;

        wg_barrier_wait();                     // all waves' stage writes visible

        // split accumulators: two independent wmma chains per result
        v8f accA, accB;
#pragma unroll
        for (int r = 0; r < 8; ++r) { accA[r] = bv; accB[r] = 0.0f; }

        // gate += x_t @ W_in   (K = 96, padded)
#pragma unroll
        for (int kb = 0; kb < 3; ++kb) {
            v16h a = load_a_frag(xS, XSTR, kb * 32, lo16, kh);
            if (kb & 1) accB = WMMA_F16(a, Bin[kb], accB);
            else        accA = WMMA_F16(a, Bin[kb], accA);
        }
        // gate += h_{t-1} @ W_rec (K = 256); same A feeds the readout of step t-1
        if (doOut) {
            v8f oaccA, oaccB;
#pragma unroll
            for (int r = 0; r < 8; ++r) { oaccA[r] = bo; oaccB[r] = 0.0f; }
#pragma unroll
            for (int kb = 0; kb < 8; ++kb) {
                v16h a  = load_a_frag(hS, HSTR, kb * 32, lo16, kh);
                v16h bf = *(const v16h*)&boutL[((size_t)tid * 8 + kb) * 16];
                if (kb & 1) { accB = WMMA_F16(a, Brec[kb], accB);
                              oaccB = WMMA_F16(a, bf, oaccB); }
                else        { accA = WMMA_F16(a, Brec[kb], accA);
                              oaccA = WMMA_F16(a, bf, oaccA); }
            }
#pragma unroll
            for (int r = 0; r < 8; ++r) {
                float yh = sigmoid_f(oaccA[r] + oaccB[r]);
                if (no < DOUT) {
                    op[(size_t)r * DOUT] = yh;
                    float e = (yv[r] - yh) * mk[r];
                    cost = fmaf(e, e, cost);
                }
            }
            yp += (size_t)B_ * DOUT;
            mp += (size_t)B_ * DOUT;
            op += (size_t)B_ * DOUT;
        } else {
#pragma unroll
            for (int kb = 0; kb < 8; ++kb) {
                v16h a = load_a_frag(hS, HSTR, kb * 32, lo16, kh);
                if (kb & 1) accB = WMMA_F16(a, Brec[kb], accB);
                else        accA = WMMA_F16(a, Brec[kb], accA);
            }
        }

        // leaky update (f32 master state lives in registers, same lane mapping)
#pragma unroll
        for (int r = 0; r < 8; ++r)
            hreg[r] = (1.0f - ALPHA) * hreg[r] + ALPHA * softplus_f(accA[r] + accB[r]);
    }

    // ---- tail: readout for s = T-1 (buffer 0 is safe: step T-1 used buffer 1) ----
    _Float16* hS = hA[0];
#pragma unroll
    for (int r = 0; r < 8; ++r)
        hS[(r + 8 * kh) * HSTR + n] = (_Float16)hreg[r];
    __syncthreads();
    if (outWave) {
        v8f oaccA, oaccB;
#pragma unroll
        for (int r = 0; r < 8; ++r) { oaccA[r] = bo; oaccB[r] = 0.0f; }
#pragma unroll
        for (int kb = 0; kb < 8; ++kb) {
            v16h a  = load_a_frag(hS, HSTR, kb * 32, lo16, kh);
            v16h bf = *(const v16h*)&boutL[((size_t)tid * 8 + kb) * 16];
            if (kb & 1) oaccB = WMMA_F16(a, bf, oaccB);
            else        oaccA = WMMA_F16(a, bf, oaccA);
        }
#pragma unroll
        for (int r = 0; r < 8; ++r) {
            float yh = sigmoid_f(oaccA[r] + oaccB[r]);
            if (no < DOUT) {
                op[(size_t)r * DOUT] = yh;
                float e = (yp[(size_t)r * DOUT] - yh) * mp[(size_t)r * DOUT];
                cost = fmaf(e, e, cost);
            }
        }
    }

    // ---- deterministic block-level cost reduction ----
    red[tid] = cost;
    __syncthreads();
#pragma unroll
    for (int off = 256; off > 0; off >>= 1) {
        if (tid < off) red[tid] += red[tid + off];
        __syncthreads();
    }
    if (tid == 0) partials[blockIdx.x] = red[0];
}

__global__ void rnn_finalize(const float* __restrict__ partials, float* __restrict__ out) {
    double s = 0.0;
    for (int i = 0; i < (B_ / 16); ++i) s += (double)partials[i];
    out[(size_t)T_ * B_ * DOUT] = (float)(s / ((double)T_ * (double)B_ * (double)DOUT));
}

extern "C" void kernel_launch(void* const* d_in, const int* in_sizes, int n_in,
                              void* d_out, int out_size, void* d_ws, size_t ws_size,
                              hipStream_t stream) {
    (void)in_sizes; (void)n_in; (void)out_size; (void)ws_size;
    const float* x      = (const float*)d_in[0];
    const float* y      = (const float*)d_in[1];
    const float* cmask  = (const float*)d_in[2];
    const float* weight = (const float*)d_in[3];
    const float* bias   = (const float*)d_in[4];
    const float* w_out  = (const float*)d_in[5];
    const float* b_out  = (const float*)d_in[6];
    float* out      = (float*)d_out;
    float* partials = (float*)d_ws;   // 32 floats

    rnn_fused<<<B_ / 16, 512, 0, stream>>>(x, y, cmask, weight, bias,
                                           w_out, b_out, out, partials);
    rnn_finalize<<<1, 1, 0, stream>>>(partials, out);
}
